// InteractionBlock_34505767256267
// MI455X (gfx1250) — matrix operand order repaired and checked by reading, not compiled, not run on hardware
//
#include <hip/hip_runtime.h>
#include <hip/hip_bf16.h>
#include <math.h>

typedef __attribute__((ext_vector_type(16))) _Float16 v16h;
typedef __attribute__((ext_vector_type(8)))  _Float16 v8h;
typedef __attribute__((ext_vector_type(4)))  _Float16 v4h;
typedef __attribute__((ext_vector_type(8)))  float    v8f;

#define F_DIM 128
#define G_DIM 50
#define LDH   136   // activation row stride (halves): 272B, 16B-aligned rows, conflict-free
#define LDW   40    // transposed-weight col stride (halves): 80B -> 20-bank step
#define TILE_R 64
#define PI_F 3.14159265358979f
#define CUT_F 10.0f

// tanh-GELU via hardware exp: 0.5*x*(1+tanh(u)) == x * sigmoid(2u) == x/(1+e^-2u)
__device__ __forceinline__ float gelu_gt2(float x) {
  float u = 0.7978845608028654f * (x + 0.044715f * x * x * x);
  return x / (1.0f + __expf(-2.0f * u));
}

__device__ __forceinline__ v8f wmma_f16(v16h a, v16h b, v8f c) {
  return __builtin_amdgcn_wmma_f32_16x16x32_f16(false, a, false, b, (short)0, c, false, false);
}

// A fragment: 16x32 f16 tile (ISA 7.12.2 layout), two contiguous b128 LDS loads.
__device__ __forceinline__ v16h load_a_frag(const _Float16* rowbase, int lane, int k0) {
  int r = lane & 15, hi = lane >> 4;
  const _Float16* p = rowbase + r * LDH + k0 + hi * 8;
  v8h lo = *(const v8h*)(p);
  v8h up = *(const v8h*)(p + 16);
  v16h a;
#pragma unroll
  for (int j = 0; j < 8; ++j) { a[j] = lo[j]; a[j + 8] = up[j]; }
  return a;
}

// B fragment: 32x16 f16 tile from transposed LDS weight wT[n][k] (stride LDW).
__device__ __forceinline__ v16h load_b_frag(const _Float16* wT, int lane, int n0) {
  int n = lane & 15, hi = lane >> 4;
  const _Float16* p = wT + (n0 + n) * LDW + hi * 8;
  v8h lo = *(const v8h*)(p);
  v8h up = *(const v8h*)(p + 16);
  v16h b;
#pragma unroll
  for (int j = 0; j < 8; ++j) { b[j] = lo[j]; b[j + 8] = up[j]; }
  return b;
}

// Stage one K-tile (32 x 128) of row-major fp32 weight into transposed f16 LDS.
// Each thread: 4 independent global_load_b128 (clause, staggered waits), then stores.
__device__ __forceinline__ void stage_wt(_Float16* wT, const float* __restrict__ Wg,
                                         int krow0, int krows, int tid) {
  int k  = tid >> 3;             // 0..31
  int nb = (tid & 7) << 4;       // col base: 0,16,...,112
  int kg = krow0 + k;
  float4 q0 = make_float4(0.f, 0.f, 0.f, 0.f), q1 = q0, q2 = q0, q3 = q0;
  if (kg < krows) {
    const float4* src = (const float4*)(Wg + (long long)kg * F_DIM + nb);
    q0 = src[0]; q1 = src[1]; q2 = src[2]; q3 = src[3];
  }
  float qs[16] = {q0.x, q0.y, q0.z, q0.w, q1.x, q1.y, q1.z, q1.w,
                  q2.x, q2.y, q2.z, q2.w, q3.x, q3.y, q3.z, q3.w};
#pragma unroll
  for (int j = 0; j < 16; ++j) wT[(nb + j) * LDW + k] = (_Float16)qs[j];
}

// Stage 64 x 128 fp32 activations into f16 LDS: float4 loads, v4h b64 stores.
__device__ __forceinline__ void stage_act(_Float16 (*dst)[LDH], const float* __restrict__ src,
                                          long long rBase, long long nrows, int tid) {
#pragma unroll
  for (int t = 0; t < 8; ++t) {
    int idx4 = tid + t * 256;          // 0..2047
    int m = idx4 >> 5;
    int k4 = (idx4 & 31) << 2;
    long long r = rBase + m;
    float4 q = make_float4(0.f, 0.f, 0.f, 0.f);
    if (r < nrows) q = *(const float4*)(src + r * F_DIM + k4);
    v4h h; h[0] = (_Float16)q.x; h[1] = (_Float16)q.y; h[2] = (_Float16)q.z; h[3] = (_Float16)q.w;
    *(v4h*)&dst[m][k4] = h;
  }
}

// ---------------------------------------------------------------------------
// xf = x @ lin1_w
// ---------------------------------------------------------------------------
__global__ __launch_bounds__(256) void lin1_kernel(const float* __restrict__ x,
                                                   const float* __restrict__ w,
                                                   float* __restrict__ xf, int N_) {
  __shared__ __align__(16) _Float16 sx[TILE_R][LDH];
  __shared__ __align__(16) _Float16 swT[2][F_DIM * LDW];
  const int tid = threadIdx.x, lane = tid & 31, wid = tid >> 5;
  const int rg = wid & 3, ch = wid >> 2;
  const long long rBase = (long long)blockIdx.x * TILE_R;

  stage_act(sx, x, rBase, N_, tid);
  stage_wt(swT[0], w, 0, F_DIM, tid);
  __syncthreads();

  v16h a[4];
#pragma unroll
  for (int kt = 0; kt < 4; ++kt) a[kt] = load_a_frag(&sx[rg * 16][0], lane, kt * 32);

  v8f acc[4] = {};
#pragma unroll
  for (int kt = 0; kt < 4; ++kt) {
    if (kt + 1 < 4) stage_wt(swT[(kt + 1) & 1], w, (kt + 1) * 32, F_DIM, tid);
#pragma unroll
    for (int i = 0; i < 4; ++i)
      acc[i] = wmma_f16(a[kt], load_b_frag(swT[kt & 1], lane, (ch * 4 + i) * 16), acc[i]);
    __syncthreads();
  }

  const int nl = lane & 15, mh = (lane >> 4) * 8;
#pragma unroll
  for (int i = 0; i < 4; ++i) {
    int n = (ch * 4 + i) * 16 + nl;
#pragma unroll
    for (int v = 0; v < 8; ++v) {
      long long r = rBase + rg * 16 + v + mh;
      if (r < N_) xf[r * F_DIM + n] = acc[i][v];
    }
  }
}

// ---------------------------------------------------------------------------
// Fused edge pipeline: 4-layer filter MLP (WMMA) + cutoff + gather/scatter-add
// ---------------------------------------------------------------------------
__global__ __launch_bounds__(256) void edge_filter_scatter_kernel(
    const float* __restrict__ edge_attr, const float* __restrict__ edge_length,
    const int* __restrict__ ei, long long E_,
    const float* __restrict__ fw0, const float* __restrict__ fb0,
    const float* __restrict__ fw1, const float* __restrict__ fb1,
    const float* __restrict__ fw2, const float* __restrict__ fb2,
    const float* __restrict__ fw3, const float* __restrict__ fb3,
    const float* __restrict__ xf, float* __restrict__ agg) {
  __shared__ __align__(16) _Float16 sh[2][TILE_R][LDH];
  __shared__ __align__(16) _Float16 swT[2][F_DIM * LDW];

  const int tid = threadIdx.x, lane = tid & 31, wid = tid >> 5;
  const int rg = wid & 3, ch = wid >> 2;
  const long long eBase = (long long)blockIdx.x * TILE_R;

  // stage edge_attr tile (flat 64*50 contiguous fp32 -> f16), vectorized
  {
    long long fBase = eBase * G_DIM;
    long long fMax = E_ * (long long)G_DIM - fBase;   // valid flat elems in tile
#pragma unroll
    for (int t = 0; t < 4; ++t) {
      int idx4 = tid + t * 256;                       // 0..1023, need < 800
      if (idx4 < (TILE_R * G_DIM) / 4) {
        int f0 = idx4 * 4;
        float4 q = make_float4(0.f, 0.f, 0.f, 0.f);
        if (f0 + 3 < fMax) q = *(const float4*)(edge_attr + fBase + f0);
        float qa[4] = {q.x, q.y, q.z, q.w};
#pragma unroll
        for (int j = 0; j < 4; ++j) {
          int f = f0 + j;
          int m = f / G_DIM;
          int k = f - G_DIM * m;
          sh[0][m][k] = (_Float16)qa[j];
        }
      }
    }
    // zero-pad K = 50..63  (64 x 14 = 896 entries)
#pragma unroll
    for (int t = 0; t < 4; ++t) {
      int idx = tid + t * 256;
      if (idx < TILE_R * (64 - G_DIM)) {
        int m = idx / (64 - G_DIM);
        int k = G_DIM + (idx - (64 - G_DIM) * m);
        sh[0][m][k] = (_Float16)0.0f;
      }
    }
  }

  const float* Ws[4] = {fw0, fw1, fw2, fw3};
  const float* Bs[4] = {fb0, fb1, fb2, fb3};
  const int nl = lane & 15, mh = (lane >> 4) * 8;

#pragma unroll
  for (int l = 0; l < 4; ++l) {
    const int Kt = (l == 0) ? 2 : 4;
    const int krows = (l == 0) ? G_DIM : F_DIM;
    const int in = (l == 0) ? 0 : ((l & 1) ? 1 : 0);   // l:0->0, 1->1, 2->0, 3->1
    const _Float16* hin = &sh[in][rg * 16][0];

    stage_wt(swT[0], Ws[l], 0, krows, tid);
    __syncthreads();                                   // fences prev epilogue + stage

    v16h a[4];
#pragma unroll
    for (int kt = 0; kt < 4; ++kt)
      if (kt < Kt) a[kt] = load_a_frag(hin, lane, kt * 32);

    v8f acc[4] = {};
#pragma unroll
    for (int kt = 0; kt < 4; ++kt) {
      if (kt < Kt) {
        if (kt + 1 < Kt) stage_wt(swT[(kt + 1) & 1], Ws[l], (kt + 1) * 32, krows, tid);
#pragma unroll
        for (int i = 0; i < 4; ++i)
          acc[i] = wmma_f16(a[kt], load_b_frag(swT[kt & 1], lane, (ch * 4 + i) * 16), acc[i]);
        __syncthreads();
      }
    }

    float bias[4];
#pragma unroll
    for (int i = 0; i < 4; ++i) bias[i] = Bs[l][(ch * 4 + i) * 16 + nl];

    if (l < 3) {
#pragma unroll
      for (int i = 0; i < 4; ++i) {
        int n = (ch * 4 + i) * 16 + nl;
#pragma unroll
        for (int v = 0; v < 8; ++v)
          sh[in ^ 1][rg * 16 + v + mh][n] = (_Float16)gelu_gt2(acc[i][v] + bias[i]);
      }
    } else {
      // W = gelu(h+b)*C(len); msg = xf[src]*W; atomicAdd into agg[dst]
#pragma unroll
      for (int v = 0; v < 8; ++v) {
        int m = rg * 16 + v + mh;
        long long e = eBase + m;
        if (e < E_) {
          float len = edge_length[e];
          float C = 0.5f * (__cosf(len * (PI_F / CUT_F)) + 1.0f);
          C = (len <= CUT_F && len >= 0.0f) ? C : 0.0f;
          int s = ei[e];
          int d = ei[E_ + e];
          const float* xrow = xf + (long long)s * F_DIM;
          float* arow = agg + (long long)d * F_DIM;
#pragma unroll
          for (int i = 0; i < 4; ++i) {
            int n = (ch * 4 + i) * 16 + nl;
            float wv = gelu_gt2(acc[i][v] + bias[i]) * C;
            atomicAdd(&arow[n], xrow[n] * wv);
          }
        }
      }
    }
  }
}

// ---------------------------------------------------------------------------
// out = gelu(agg @ lin2_w + b2) @ lin_w + b   (two WMMA GEMMs fused via LDS)
// ---------------------------------------------------------------------------
__global__ __launch_bounds__(256) void out_kernel(
    const float* __restrict__ agg, const float* __restrict__ w2, const float* __restrict__ b2,
    const float* __restrict__ w3, const float* __restrict__ b3,
    float* __restrict__ out, int N_) {
  __shared__ __align__(16) _Float16 sA[TILE_R][LDH];
  __shared__ __align__(16) _Float16 sM[TILE_R][LDH];
  __shared__ __align__(16) _Float16 swT[2][F_DIM * LDW];
  const int tid = threadIdx.x, lane = tid & 31, wid = tid >> 5;
  const int rg = wid & 3, ch = wid >> 2;
  const long long rBase = (long long)blockIdx.x * TILE_R;
  const int nl = lane & 15, mh = (lane >> 4) * 8;

  stage_act(sA, agg, rBase, N_, tid);

  // GEMM 1: agg @ lin2_w + b2, gelu -> sM
  {
    stage_wt(swT[0], w2, 0, F_DIM, tid);
    __syncthreads();
    v16h a[4];
#pragma unroll
    for (int kt = 0; kt < 4; ++kt) a[kt] = load_a_frag(&sA[rg * 16][0], lane, kt * 32);
    v8f acc[4] = {};
#pragma unroll
    for (int kt = 0; kt < 4; ++kt) {
      if (kt + 1 < 4) stage_wt(swT[(kt + 1) & 1], w2, (kt + 1) * 32, F_DIM, tid);
#pragma unroll
      for (int i = 0; i < 4; ++i)
        acc[i] = wmma_f16(a[kt], load_b_frag(swT[kt & 1], lane, (ch * 4 + i) * 16), acc[i]);
      __syncthreads();
    }
#pragma unroll
    for (int i = 0; i < 4; ++i) {
      int n = (ch * 4 + i) * 16 + nl;
      float bias = b2[n];
#pragma unroll
      for (int v = 0; v < 8; ++v)
        sM[rg * 16 + v + mh][n] = (_Float16)gelu_gt2(acc[i][v] + bias);
    }
  }
  // GEMM 2: sM @ lin_w + b -> out
  {
    stage_wt(swT[0], w3, 0, F_DIM, tid);
    __syncthreads();                      // fences sM stores
    v16h a[4];
#pragma unroll
    for (int kt = 0; kt < 4; ++kt) a[kt] = load_a_frag(&sM[rg * 16][0], lane, kt * 32);
    v8f acc[4] = {};
#pragma unroll
    for (int kt = 0; kt < 4; ++kt) {
      if (kt + 1 < 4) stage_wt(swT[(kt + 1) & 1], w3, (kt + 1) * 32, F_DIM, tid);
#pragma unroll
      for (int i = 0; i < 4; ++i)
        acc[i] = wmma_f16(a[kt], load_b_frag(swT[kt & 1], lane, (ch * 4 + i) * 16), acc[i]);
      __syncthreads();
    }
#pragma unroll
    for (int i = 0; i < 4; ++i) {
      int n = (ch * 4 + i) * 16 + nl;
      float bias = b3[n];
#pragma unroll
      for (int v = 0; v < 8; ++v) {
        long long r = rBase + rg * 16 + v + mh;
        if (r < N_) out[r * F_DIM + n] = acc[i][v] + bias;
      }
    }
  }
}

__global__ __launch_bounds__(256) void zero_kernel(float* __restrict__ p, long long n) {
  long long i = (long long)blockIdx.x * blockDim.x + threadIdx.x;
  long long stride = (long long)gridDim.x * blockDim.x;
  for (; i < n; i += stride) p[i] = 0.0f;
}

extern "C" void kernel_launch(void* const* d_in, const int* in_sizes, int n_in,
                              void* d_out, int out_size, void* d_ws, size_t ws_size,
                              hipStream_t stream) {
  const float* x     = (const float*)d_in[0];
  const int*   ei    = (const int*)  d_in[1];
  const float* elen  = (const float*)d_in[2];
  const float* eattr = (const float*)d_in[3];
  const float* fw0 = (const float*)d_in[4];  const float* fb0 = (const float*)d_in[5];
  const float* fw1 = (const float*)d_in[6];  const float* fb1 = (const float*)d_in[7];
  const float* fw2 = (const float*)d_in[8];  const float* fb2 = (const float*)d_in[9];
  const float* fw3 = (const float*)d_in[10]; const float* fb3 = (const float*)d_in[11];
  const float* lin1w = (const float*)d_in[12];
  const float* lin2w = (const float*)d_in[13]; const float* lin2b = (const float*)d_in[14];
  const float* linw  = (const float*)d_in[15]; const float* linb  = (const float*)d_in[16];
  float* out = (float*)d_out;

  const int N_       = in_sizes[0] / F_DIM;       // 40000
  const long long E_ = (long long)in_sizes[2];    // 600000

  float* xf  = (float*)d_ws;                      // N x 128 fp32
  float* agg = xf + (size_t)N_ * F_DIM;           // N x 128 fp32

  const int nb = (N_ + TILE_R - 1) / TILE_R;
  const int eb = (int)((E_ + TILE_R - 1) / TILE_R);

  zero_kernel<<<1024, 256, 0, stream>>>(agg, (long long)N_ * F_DIM);
  lin1_kernel<<<nb, 256, 0, stream>>>(x, lin1w, xf, N_);
  edge_filter_scatter_kernel<<<eb, 256, 0, stream>>>(
      eattr, elen, ei, E_, fw0, fb0, fw1, fb1, fw2, fb2, fw3, fb3, xf, agg);
  out_kernel<<<nb, 256, 0, stream>>>(agg, lin2w, lin2b, linw, linb, out, N_);
}
